// AMSPNet_fixed_iCOH_4183298146528
// MI455X (gfx1250) — matrix operand order repaired and checked by reading, not compiled, not run on hardware
//
#include <hip/hip_runtime.h>
#include <hip/hip_bf16.h>
#include <math.h>

typedef _Float16 half_t;
typedef __attribute__((ext_vector_type(16))) _Float16 v16h;
typedef __attribute__((ext_vector_type(8)))  _Float16 v8h;
typedef __attribute__((ext_vector_type(8)))  float    v8f;

#define BSZ   1024
#define CCH   21
#define BC    (BSZ*CCH)      /* 21504 */
#define WINL  256
#define HWIN  128
#define LSIG  512
#define SSEG  5
#define HID   315
#define NGCN  128
#define KGCN  256
#define KPAD  448            /* 443 padded */
#define NRU   640            /* 630 padded */
#define NCQ   320            /* 315 padded */
#define TWO_PI_OVER_N (6.283185307179586f / 256.0f)

__device__ __forceinline__ float sigmoidf_(float v) { return 1.0f / (1.0f + expf(-v)); }

// ---------------------------------------------------------------------------
// set_fft: x (B,1,C,1024) -> ft (B,C,512). One block per 256-sample window.
// ---------------------------------------------------------------------------
__global__ void k_setfft(const float* __restrict__ x, float* __restrict__ ft) {
  __shared__ float xs[WINL];
  __shared__ float tw[WINL];
  const int bid = blockIdx.x;                 // (b*C + c)*4 + w
  const int t = threadIdx.x;                  // 0..127
  const float* xp = x + (size_t)bid * WINL;
  xs[t] = xp[t];
  xs[t + HWIN] = xp[t + HWIN];
  tw[t] = cosf(TWO_PI_OVER_N * (float)t);
  tw[t + HWIN] = cosf(TWO_PI_OVER_N * (float)(t + HWIN));
  __syncthreads();
  const int k = t;
  float sr = 0.f, si = 0.f;
  for (int n = 0; n < WINL; ++n) {
    const int ph = (k * n) & 255;
    const float xv = xs[n];
    sr += xv * tw[ph];                        // cos
    si -= xv * tw[(ph + 192) & 255];          // sin(th) = cos(th + 3pi/2)
  }
  float amp = sqrtf(sr * sr + si * si);
  if (amp == 0.f) amp = 1e-8f;
  ft[(size_t)(bid >> 2) * LSIG + (size_t)(bid & 3) * HWIN + k] = logf(amp);
}

// ---------------------------------------------------------------------------
// Full 256-pt DFT of segment window s for every (b,c): re/im/mag (BC,256)
// ---------------------------------------------------------------------------
__global__ void k_dft(const float* __restrict__ ft, float* __restrict__ re,
                      float* __restrict__ im, float* __restrict__ mg, int s) {
  __shared__ float xs[WINL];
  __shared__ float tw[WINL];
  const int bc = blockIdx.x;
  const int k = threadIdx.x;                  // 0..255
  xs[k] = ft[(size_t)bc * LSIG + s * 64 + k];
  tw[k] = cosf(TWO_PI_OVER_N * (float)k);
  __syncthreads();
  float sr = 0.f, si = 0.f;
  for (int n = 0; n < WINL; ++n) {
    const int ph = (k * n) & 255;
    const float xv = xs[n];
    sr += xv * tw[ph];
    si -= xv * tw[(ph + 192) & 255];
  }
  const size_t o = (size_t)bc * WINL + k;
  re[o] = sr; im[o] = si; mg[o] = sqrtf(sr * sr + si * si);
}

// ---------------------------------------------------------------------------
// icoh Gram: R[b,s,i,j] = (sum im_i re_j - sum re_i im_j) / (sum |F_i||F_j|)
// zero diagonal. One thread per (b,i,j).
// ---------------------------------------------------------------------------
__global__ void k_gram(const float* __restrict__ re, const float* __restrict__ im,
                       const float* __restrict__ mg, float* __restrict__ R, int s) {
  const int tid = blockIdx.x * blockDim.x + threadIdx.x;
  if (tid >= BSZ * 441) return;
  const int b = tid / 441;
  const int ij = tid - b * 441;
  const int i = ij / CCH, j = ij - i * CCH;
  const float* rei = re + ((size_t)b * CCH + i) * WINL;
  const float* imi = im + ((size_t)b * CCH + i) * WINL;
  const float* rej = re + ((size_t)b * CCH + j) * WINL;
  const float* imj = im + ((size_t)b * CCH + j) * WINL;
  const float* mgi = mg + ((size_t)b * CCH + i) * WINL;
  const float* mgj = mg + ((size_t)b * CCH + j) * WINL;
  float n1 = 0.f, n2 = 0.f, dd = 0.f;
  for (int f = 0; f < WINL; ++f) {
    n1 += imi[f] * rej[f];
    n2 += rei[f] * imj[f];
    dd += mgi[f] * mgj[f];
  }
  R[((size_t)b * SSEG + s) * 441 + ij] = (i == j) ? 0.f : (n1 - n2) / dd;
}

// ---------------------------------------------------------------------------
// Build f16 A matrix for GCN GEMM: Axt[b*21+c][k] = ft[b,c,s*64+k]
// ---------------------------------------------------------------------------
__global__ void k_xt_half(const float* __restrict__ ft, half_t* __restrict__ Axt, int s) {
  const size_t idx = (size_t)blockIdx.x * blockDim.x + threadIdx.x;
  if (idx >= (size_t)BC * WINL) return;
  const int k = (int)(idx & 255);
  const size_t bc = idx >> 8;
  Axt[idx] = (half_t)ft[bc * LSIG + s * 64 + k];
}

// ---------------------------------------------------------------------------
// WMMA f16 GEMM: C[M x N] = A[M x K](f16,row) * Bt[N x K](f16)^T + bias[N]
// M = gridDim.x*128 (exact), N = gridDim.y*64 (exact), K % 32 == 0.
// Block = 256 thr = 8 waves; wave tile = 16M x 64N (4 accumulators).
// ---------------------------------------------------------------------------
__global__ __launch_bounds__(256) void k_gemm_wmma(
    const half_t* __restrict__ A, const half_t* __restrict__ Bt,
    const float* __restrict__ bias, float* __restrict__ C, int N, int K) {
  const int wave = threadIdx.x >> 5;
  const int lane = threadIdx.x & 31;
  const int hi   = lane >> 4;                 // 0/1: which K sub-block this lane feeds
  const int r    = lane & 15;
  const int m0   = blockIdx.x * 128 + wave * 16;
  const int n0   = blockIdx.y * 64;

  v8f acc0 = {0.f,0.f,0.f,0.f,0.f,0.f,0.f,0.f};
  v8f acc1 = acc0, acc2 = acc0, acc3 = acc0;

  // A frag: lane-lo holds K {k0..k0+15}; lane-hi holds K {k0+16..k0+31}.
  const half_t* Ap  = A  + (size_t)(m0 + r) * K + hi * 16;
  // B frag: lane-lo holds K {k0..k0+15}; lane-hi holds K {k0+16..k0+31}.
  const half_t* Bp0 = Bt + (size_t)(n0 +  0 + r) * K + hi * 16;
  const half_t* Bp1 = Bt + (size_t)(n0 + 16 + r) * K + hi * 16;
  const half_t* Bp2 = Bt + (size_t)(n0 + 32 + r) * K + hi * 16;
  const half_t* Bp3 = Bt + (size_t)(n0 + 48 + r) * K + hi * 16;

  for (int k0 = 0; k0 < K; k0 += 32) {
    v16h a, b0, b1, b2, b3;
    ((v8h*)&a)[0]  = *(const v8h*)(Ap  + k0);
    ((v8h*)&a)[1]  = *(const v8h*)(Ap  + k0 + 8);
    ((v8h*)&b0)[0] = *(const v8h*)(Bp0 + k0);
    ((v8h*)&b0)[1] = *(const v8h*)(Bp0 + k0 + 8);
    ((v8h*)&b1)[0] = *(const v8h*)(Bp1 + k0);
    ((v8h*)&b1)[1] = *(const v8h*)(Bp1 + k0 + 8);
    ((v8h*)&b2)[0] = *(const v8h*)(Bp2 + k0);
    ((v8h*)&b2)[1] = *(const v8h*)(Bp2 + k0 + 8);
    ((v8h*)&b3)[0] = *(const v8h*)(Bp3 + k0);
    ((v8h*)&b3)[1] = *(const v8h*)(Bp3 + k0 + 8);
    acc0 = __builtin_amdgcn_wmma_f32_16x16x32_f16(false, a, false, b0, (short)0, acc0, false, false);
    acc1 = __builtin_amdgcn_wmma_f32_16x16x32_f16(false, a, false, b1, (short)0, acc1, false, false);
    acc2 = __builtin_amdgcn_wmma_f32_16x16x32_f16(false, a, false, b2, (short)0, acc2, false, false);
    acc3 = __builtin_amdgcn_wmma_f32_16x16x32_f16(false, a, false, b3, (short)0, acc3, false, false);
  }

  const float bv0 = bias ? bias[n0 +  0 + r] : 0.f;
  const float bv1 = bias ? bias[n0 + 16 + r] : 0.f;
  const float bv2 = bias ? bias[n0 + 32 + r] : 0.f;
  const float bv3 = bias ? bias[n0 + 48 + r] : 0.f;
  // D layout: lane<16 -> c[v] = C[m0+v][n0+..+r]; lane>=16 -> row +8
  for (int v = 0; v < 8; ++v) {
    const size_t ro = (size_t)(m0 + hi * 8 + v) * N;
    C[ro + n0 +  0 + r] = acc0[v] + bv0;
    C[ro + n0 + 16 + r] = acc1[v] + bv1;
    C[ro + n0 + 32 + r] = acc2[v] + bv2;
    C[ro + n0 + 48 + r] = acc3[v] + bv3;
  }
}

// ---------------------------------------------------------------------------
// g = sigmoid(R) @ gcnpre  (per batch 21x21 @ 21x128). One block per (b,i).
// ---------------------------------------------------------------------------
__global__ void k_gcn_apply(const float* __restrict__ R, const float* __restrict__ gcnpre,
                            float* __restrict__ g, int s) {
  __shared__ float arow[CCH];
  const int bc = blockIdx.x;
  const int b = bc / CCH, i = bc - b * CCH;
  if (threadIdx.x < CCH)
    arow[threadIdx.x] = sigmoidf_(R[((size_t)b * SSEG + s) * 441 + i * CCH + threadIdx.x]);
  __syncthreads();
  const int n = threadIdx.x;                  // 0..127
  float acc = 0.f;
  for (int j = 0; j < CCH; ++j)
    acc += arow[j] * gcnpre[((size_t)b * CCH + j) * NGCN + n];
  g[(size_t)bc * NGCN + n] = acc;
}

// cat f16 (BC x 448): [g | h | 0-pad]
__global__ void k_build_cat(const float* __restrict__ g, const float* __restrict__ h,
                            half_t* __restrict__ cat) {
  const size_t idx = (size_t)blockIdx.x * blockDim.x + threadIdx.x;
  if (idx >= (size_t)BC * KPAD) return;
  const int col = (int)(idx % KPAD);
  const size_t row = idx / KPAD;
  float v = 0.f;
  if (col < NGCN)            v = g[row * NGCN + col];
  else if (col < NGCN + HID) v = h[row * HID + (col - NGCN)];
  cat[idx] = (half_t)v;
}

// r = sigmoid(ru[:, :315]); overwrite cat h-half with (f16)(r*h)
__global__ void k_gates(const float* __restrict__ ru, const float* __restrict__ h,
                        half_t* __restrict__ cat) {
  const size_t idx = (size_t)blockIdx.x * blockDim.x + threadIdx.x;
  if (idx >= (size_t)BC * HID) return;
  const size_t row = idx / HID;
  const int k = (int)(idx - row * HID);
  const float rr = sigmoidf_(ru[row * NRU + k]);
  cat[row * KPAD + NGCN + k] = (half_t)(rr * h[idx]);
}

// h = u*h + (1-u)*tanh(cpre)   (u = sigmoid(ru[:, 315:630]))
__global__ void k_hnew(const float* __restrict__ ru, const float* __restrict__ cpre,
                       float* __restrict__ h) {
  const size_t idx = (size_t)blockIdx.x * blockDim.x + threadIdx.x;
  if (idx >= (size_t)BC * HID) return;
  const size_t row = idx / HID;
  const int k = (int)(idx - row * HID);
  const float u = sigmoidf_(ru[row * NRU + HID + k]);
  const float c = tanhf(cpre[row * NCQ + k]);
  h[idx] = u * h[idx] + (1.f - u) * c;
}

__global__ void k_zero(float* __restrict__ p, int n) {
  const int i = blockIdx.x * blockDim.x + threadIdx.x;
  if (i < n) p[i] = 0.f;
}

// One-shot weight prep: transpose + f16-convert + pad, padded biases.
__global__ void k_prep(const float* __restrict__ gcnW, const float* __restrict__ Wr,
                       const float* __restrict__ Wu, const float* __restrict__ Wc,
                       const float* __restrict__ br, const float* __restrict__ bu,
                       const float* __restrict__ bcv,
                       half_t* __restrict__ gcnWt, half_t* __restrict__ Wrut,
                       half_t* __restrict__ Wct, float* __restrict__ brupad,
                       float* __restrict__ bcpad) {
  const int idx = blockIdx.x * blockDim.x + threadIdx.x;
  if (idx < NGCN * KGCN) {                    // gcnWt: (128 x 256), gcn_W is (256,128)
    const int n = idx / KGCN, k = idx - n * KGCN;
    gcnWt[idx] = (half_t)gcnW[k * NGCN + n];
  }
  if (idx < NRU * KPAD) {                     // Wrut: (640 x 448), W_r/W_u are (443,315)
    const int n = idx / KPAD, k = idx - n * KPAD;
    float v = 0.f;
    if (k < 443) {
      if (n < HID)            v = Wr[k * HID + n];
      else if (n < 2 * HID)   v = Wu[k * HID + (n - HID)];
    }
    Wrut[idx] = (half_t)v;
  }
  if (idx < NCQ * KPAD) {                     // Wct: (320 x 448), W_c is (443,315)
    const int n = idx / KPAD, k = idx - n * KPAD;
    float v = 0.f;
    if (k < 443 && n < HID) v = Wc[k * HID + n];
    Wct[idx] = (half_t)v;
  }
  if (idx < NRU) brupad[idx] = (idx < HID) ? br[idx] : ((idx < 2 * HID) ? bu[idx - HID] : 0.f);
  if (idx < NCQ) bcpad[idx] = (idx < HID) ? bcv[idx] : 0.f;
}

// ---------------------------------------------------------------------------
// Spatial branch: tiny 3x3 attention per (s, comb, b)
// ---------------------------------------------------------------------------
__device__ __forceinline__ void mat3_lin(const float X[3][3], const float* W,
                                         const float* bb, float Y[3][3]) {
  for (int p = 0; p < 3; ++p)
    for (int q = 0; q < 3; ++q) {
      float acc = bb[q];
      for (int j = 0; j < 3; ++j) acc += X[p][j] * W[j * 3 + q];
      Y[p][q] = acc;
    }
}

__device__ __forceinline__ void attn3(const float Q[3][3], const float Km[3][3],
                                      const float V[3][3], float O[3][3]) {
  const float inv = 0.5773502691896258f;      // 1/sqrt(3)
  for (int p = 0; p < 3; ++p) {
    float sr[3];
    for (int q = 0; q < 3; ++q) {
      float acc = 0.f;
      for (int j = 0; j < 3; ++j) acc += Q[p][j] * Km[q][j];
      sr[q] = acc * inv;
    }
    const float m = fmaxf(sr[0], fmaxf(sr[1], sr[2]));
    const float e0 = expf(sr[0] - m), e1 = expf(sr[1] - m), e2 = expf(sr[2] - m);
    const float inv_s = 1.f / (e0 + e1 + e2);
    for (int q = 0; q < 3; ++q)
      O[p][q] = (e0 * V[0][q] + e1 * V[1][q] + e2 * V[2][q]) * inv_s;
  }
}

__global__ void k_attn(const float* __restrict__ R, const float* __restrict__ mW,
                       const float* __restrict__ mb, float* __restrict__ ao) {
  const int tid = blockIdx.x * blockDim.x + threadIdx.x;
  if (tid >= SSEG * 7 * BSZ) return;
  const int b = tid & 1023;
  const int sc = tid >> 10;                   // s*7 + comb
  const int comb = sc % 7, s = sc / 7;
  const int CMB[7][3] = {{0,1,2},{1,0,4},{2,0,5},{3,2,4},{4,1,6},{5,3,6},{6,3,5}};
  const float* Rb = R + ((size_t)b * SSEG + s) * 441;
  float X[3][3][3];
  for (int m = 0; m < 3; ++m) {
    const int g3 = 3 * CMB[comb][m];
    for (int p = 0; p < 3; ++p)
      for (int q = 0; q < 3; ++q)
        X[m][p][q] = (p == q) ? 0.f : Rb[(g3 + p) * CCH + (g3 + q)];
  }
  float Qa[3][3], Ka[3][3], Va[3][3], Qb[3][3], Kb[3][3], Vb[3][3], Qc[3][3], Kc[3][3], Vc[3][3];
  mat3_lin(X[0], mW + 0 * 9, mb + 0 * 3, Qa);
  mat3_lin(X[0], mW + 1 * 9, mb + 1 * 3, Ka);
  mat3_lin(X[0], mW + 2 * 9, mb + 2 * 3, Va);
  mat3_lin(X[1], mW + 3 * 9, mb + 3 * 3, Qb);
  mat3_lin(X[1], mW + 4 * 9, mb + 4 * 3, Kb);
  mat3_lin(X[1], mW + 5 * 9, mb + 5 * 3, Vb);
  mat3_lin(X[2], mW + 6 * 9, mb + 6 * 3, Qc);
  mat3_lin(X[2], mW + 7 * 9, mb + 7 * 3, Kc);
  mat3_lin(X[2], mW + 8 * 9, mb + 8 * 3, Vc);
  float O0[3][3], O1[3][3], O2[3][3];
  attn3(Qc, Ka, Vb, O0);
  attn3(Qa, Kb, Vc, O1);
  attn3(Qb, Kc, Va, O2);
  float* dst = ao + ((size_t)sc * BSZ + b) * 27;
  for (int p = 0; p < 3; ++p)
    for (int q = 0; q < 3; ++q) {
      dst[p * 9 + 0 + q] = O0[p][q];
      dst[p * 9 + 3 + q] = O1[p][q];
      dst[p * 9 + 6 + q] = O2[p][q];
    }
}

// BatchNorm stats over B*3 = 3072 samples per (s,comb,j). Block per (s*7+comb)*9+j.
__global__ void k_bnstats(const float* __restrict__ ao, float* __restrict__ mu,
                          float* __restrict__ var) {
  __shared__ float s1[256], s2[256];
  const int scj = blockIdx.x;                 // 0..314
  const int j = scj % 9, sc = scj / 9;
  float a = 0.f, b2 = 0.f;
  for (int i = threadIdx.x; i < 3 * BSZ; i += 256) {
    const int b = i / 3, p = i - b * 3;
    const float v = ao[((size_t)sc * BSZ + b) * 27 + p * 9 + j];
    a += v; b2 += v * v;
  }
  s1[threadIdx.x] = a; s2[threadIdx.x] = b2;
  __syncthreads();
  for (int st = 128; st > 0; st >>= 1) {
    if (threadIdx.x < st) { s1[threadIdx.x] += s1[threadIdx.x + st]; s2[threadIdx.x] += s2[threadIdx.x + st]; }
    __syncthreads();
  }
  if (threadIdx.x == 0) {
    const float m = s1[0] / 3072.f;
    mu[scj] = m;
    var[scj] = s2[0] / 3072.f - m * m;
  }
}

// normalize + tiny FC(9->3) + scatter into out_spa (B, 315)
__global__ void k_spaout(const float* __restrict__ ao, const float* __restrict__ mu,
                         const float* __restrict__ var, const float* __restrict__ gamma,
                         const float* __restrict__ beta, const float* __restrict__ fcW,
                         const float* __restrict__ fcb, float* __restrict__ out_spa) {
  const int tid = blockIdx.x * blockDim.x + threadIdx.x;
  if (tid >= SSEG * 7 * BSZ) return;
  const int b = tid & 1023;
  const int sc = tid >> 10;
  const int comb = sc % 7, s = sc / 7;
  const float* src = ao + ((size_t)sc * BSZ + b) * 27;
  float nrm[3][9];
  for (int p = 0; p < 3; ++p)
    for (int j = 0; j < 9; ++j) {
      const int scj = sc * 9 + j;
      nrm[p][j] = (src[p * 9 + j] - mu[scj]) * rsqrtf(var[scj] + 1e-5f) * gamma[j] + beta[j];
    }
  for (int p = 0; p < 3; ++p)
    for (int q = 0; q < 3; ++q) {
      float acc = fcb[q];
      for (int j = 0; j < 9; ++j) acc += nrm[p][j] * fcW[j * 3 + q];
      out_spa[(size_t)b * HID + ((comb * 3 + p) * SSEG + s) * 3 + q] = acc;
    }
}

// final: attention-fuse temporal/spatial, FC 315->21->3. One thread per b.
__global__ void k_final(const float* __restrict__ h, const float* __restrict__ out_spa,
                        const float* __restrict__ awt, const float* __restrict__ aws,
                        const float* __restrict__ fc1W, const float* __restrict__ fc1b,
                        const float* __restrict__ fc2W, const float* __restrict__ fc2b,
                        float* __restrict__ out) {
  const int b = blockIdx.x * blockDim.x + threadIdx.x;
  if (b >= BSZ) return;
  const float* tem = h + ((size_t)b * CCH + (CCH - 1)) * HID;   // channel 20
  const float* spa = out_spa + (size_t)b * HID;
  float s1 = 0.f, s2 = 0.f;
  for (int k = 0; k < HID; ++k) { s1 += tem[k] * awt[k]; s2 += spa[k] * aws[k]; }
  const float m = fmaxf(s1, s2);
  const float e1 = expf(s1 - m), e2 = expf(s2 - m);
  const float w1 = e1 / (e1 + e2), w2 = e2 / (e1 + e2);
  float o0 = fc2b[0], o1 = fc2b[1], o2 = fc2b[2];
  for (int c = 0; c < CCH; ++c) {
    float acc = fc1b[c];
    for (int k = 0; k < HID; ++k) {
      const float conf = w1 * tem[k] + w2 * spa[k];
      acc += conf * fc1W[k * CCH + c];
    }
    o0 += acc * fc2W[c * 3 + 0];
    o1 += acc * fc2W[c * 3 + 1];
    o2 += acc * fc2W[c * 3 + 2];
  }
  out[(size_t)b * 3 + 0] = o0;
  out[(size_t)b * 3 + 1] = o1;
  out[(size_t)b * 3 + 2] = o2;
}

// ---------------------------------------------------------------------------
extern "C" void kernel_launch(void* const* d_in, const int* in_sizes, int n_in,
                              void* d_out, int out_size, void* d_ws, size_t ws_size,
                              hipStream_t stream) {
  (void)in_sizes; (void)n_in; (void)out_size; (void)ws_size;
  const float* x     = (const float*)d_in[0];
  const float* mW    = (const float*)d_in[1];
  const float* mb_   = (const float*)d_in[2];
  const float* bn_g  = (const float*)d_in[3];
  const float* bn_b  = (const float*)d_in[4];
  const float* mfcW  = (const float*)d_in[5];
  const float* mfcb  = (const float*)d_in[6];
  const float* gcnW  = (const float*)d_in[7];
  const float* gcnb  = (const float*)d_in[8];
  const float* Wr    = (const float*)d_in[9];
  const float* br    = (const float*)d_in[10];
  const float* Wu    = (const float*)d_in[11];
  const float* bu    = (const float*)d_in[12];
  const float* Wc    = (const float*)d_in[13];
  const float* bc    = (const float*)d_in[14];
  const float* awt   = (const float*)d_in[15];
  const float* aws   = (const float*)d_in[16];
  const float* fc1W  = (const float*)d_in[17];
  const float* fc1b  = (const float*)d_in[18];
  const float* fc2W  = (const float*)d_in[19];
  const float* fc2b  = (const float*)d_in[20];
  float* out = (float*)d_out;

  char* ws = (char*)d_ws;
  size_t off = 0;
  auto alloc = [&](size_t bytes) -> void* {
    void* p = (void*)(ws + off);
    off += (bytes + 255) & ~(size_t)255;
    return p;
  };
  float*  ft     = (float*) alloc((size_t)BC * LSIG * 4);
  float*  R      = (float*) alloc((size_t)BSZ * SSEG * 441 * 4);
  float*  re     = (float*) alloc((size_t)BC * WINL * 4);
  float*  im     = (float*) alloc((size_t)BC * WINL * 4);
  float*  mg     = (float*) alloc((size_t)BC * WINL * 4);
  half_t* Axt    = (half_t*)alloc((size_t)BC * KGCN * 2);
  half_t* gcnWt  = (half_t*)alloc((size_t)NGCN * KGCN * 2);
  half_t* Wrut   = (half_t*)alloc((size_t)NRU * KPAD * 2);
  half_t* Wct    = (half_t*)alloc((size_t)NCQ * KPAD * 2);
  float*  brupad = (float*) alloc((size_t)NRU * 4);
  float*  bcpad  = (float*) alloc((size_t)NCQ * 4);
  float*  gcnpre = (float*) alloc((size_t)BC * NGCN * 4);
  float*  g      = (float*) alloc((size_t)BC * NGCN * 4);
  half_t* cat    = (half_t*)alloc((size_t)BC * KPAD * 2);
  float*  ru     = (float*) alloc((size_t)BC * NRU * 4);
  float*  cpre   = (float*) alloc((size_t)BC * NCQ * 4);
  float*  h      = (float*) alloc((size_t)BC * HID * 4);
  float*  ao     = (float*) alloc((size_t)SSEG * 7 * BSZ * 27 * 4);
  float*  mu     = (float*) alloc((size_t)315 * 4);
  float*  var    = (float*) alloc((size_t)315 * 4);
  float*  o_spa  = (float*) alloc((size_t)BSZ * HID * 4);

  // ---- frontend: log-amplitude spectrogram ----
  k_setfft<<<BC * 4, 128, 0, stream>>>(x, ft);
  // ---- weight prep (deterministic each call) ----
  k_prep<<<(NRU * KPAD + 255) / 256, 256, 0, stream>>>(gcnW, Wr, Wu, Wc, br, bu, bc,
                                                       gcnWt, Wrut, Wct, brupad, bcpad);
  k_zero<<<(BC * HID + 255) / 256, 256, 0, stream>>>(h, BC * HID);

  // ---- GRU over 5 segments ----
  for (int s = 0; s < SSEG; ++s) {
    k_dft<<<BC, 256, 0, stream>>>(ft, re, im, mg, s);
    k_gram<<<(BSZ * 441 + 255) / 256, 256, 0, stream>>>(re, im, mg, R, s);
    k_xt_half<<<(BC * WINL + 255) / 256, 256, 0, stream>>>(ft, Axt, s);
    k_gemm_wmma<<<dim3(BC / 128, NGCN / 64), 256, 0, stream>>>(Axt, gcnWt, gcnb, gcnpre, NGCN, KGCN);
    k_gcn_apply<<<BC, 128, 0, stream>>>(R, gcnpre, g, s);
    k_build_cat<<<(BC * KPAD + 255) / 256, 256, 0, stream>>>(g, h, cat);
    k_gemm_wmma<<<dim3(BC / 128, NRU / 64), 256, 0, stream>>>(cat, Wrut, brupad, ru, NRU, KPAD);
    k_gates<<<(BC * HID + 255) / 256, 256, 0, stream>>>(ru, h, cat);
    k_gemm_wmma<<<dim3(BC / 128, NCQ / 64), 256, 0, stream>>>(cat, Wct, bcpad, cpre, NCQ, KPAD);
    k_hnew<<<(BC * HID + 255) / 256, 256, 0, stream>>>(ru, cpre, h);
  }

  // ---- spatial branch (reuses R diagonal 3x3 blocks) ----
  k_attn<<<(SSEG * 7 * BSZ + 255) / 256, 256, 0, stream>>>(R, mW, mb_, ao);
  k_bnstats<<<315, 256, 0, stream>>>(ao, mu, var);
  k_spaout<<<(SSEG * 7 * BSZ + 255) / 256, 256, 0, stream>>>(ao, mu, var, bn_g, bn_b,
                                                             mfcW, mfcb, o_spa);
  // ---- fusion + classifier ----
  k_final<<<(BSZ + 63) / 64, 64, 0, stream>>>(h, o_spa, awt, aws, fc1W, fc1b, fc2W, fc2b, out);
}
